// EquivariantUpdate_71846212927515
// MI455X (gfx1250) — matrix operand order repaired and checked by reading, not compiled, not run on hardware
//
#include <hip/hip_runtime.h>

typedef __attribute__((ext_vector_type(16))) _Float16 v16h;
typedef __attribute__((ext_vector_type(8)))  float    v8f;

#define N_NODES 100000
#define N_EDGES 800000
#define HNF 128
#define IN_EDGE 257

// workspace layout (bytes):
//   [0,      65536)   W1f : packed f16 B-fragments, 8kt x 8nt x (32 lanes x 16 halfs)
//   [65536,  98304)   W2f : packed f16 B-fragments, 4kt x 8nt x (32 lanes x 16 halfs)
//   [98304,  98304 + N_NODES*3*4)  agg : float accumulation buffer (zeroed per launch)
#define WS_W1F 0
#define WS_W2F 65536
#define WS_AGG 98304

__device__ __forceinline__ float silu_f(float x) {
    return x / (1.0f + __expf(-x));
}

#define WMMA_F16(A, B, C) \
    __builtin_amdgcn_wmma_f32_16x16x32_f16(false, (A), false, (B), (short)0, (C), false, false)

// Build one 16x32 f16 A-fragment for layer 1: lane holds row m of the
// concatenated [h[row] | h[col]] feature vector, K runs {f0..f0+7, f0+16..f0+23}.
// Runs are 8-aligned and never cross the 128-feature boundary.
__device__ __forceinline__ v16h build_a_frag(const float* __restrict__ hrow,
                                             const float* __restrict__ hcol,
                                             int f0) {
    int f1 = f0 + 16;
    const float* s0 = (f0 < HNF) ? (hrow + f0) : (hcol + (f0 - HNF));
    const float* s1 = (f1 < HNF) ? (hrow + f1) : (hcol + (f1 - HNF));
    float4 p0 = ((const float4*)s0)[0];
    float4 p1 = ((const float4*)s0)[1];
    float4 q0 = ((const float4*)s1)[0];
    float4 q1 = ((const float4*)s1)[1];
    v16h a;
    a[0]  = (_Float16)p0.x; a[1]  = (_Float16)p0.y;
    a[2]  = (_Float16)p0.z; a[3]  = (_Float16)p0.w;
    a[4]  = (_Float16)p1.x; a[5]  = (_Float16)p1.y;
    a[6]  = (_Float16)p1.z; a[7]  = (_Float16)p1.w;
    a[8]  = (_Float16)q0.x; a[9]  = (_Float16)q0.y;
    a[10] = (_Float16)q0.z; a[11] = (_Float16)q0.w;
    a[12] = (_Float16)q1.x; a[13] = (_Float16)q1.y;
    a[14] = (_Float16)q1.z; a[15] = (_Float16)q1.w;
    return a;
}

// Build one 16x32 f16 A-fragment for layer 2 from LDS activations (already f16).
__device__ __forceinline__ v16h build_a_frag_lds(const _Float16* __restrict__ arow,
                                                 int f0) {
    float4 p = *(const float4*)(arow + f0);        // 8 halfs
    float4 q = *(const float4*)(arow + f0 + 16);   // 8 halfs
    const _Float16* ph = (const _Float16*)&p;
    const _Float16* qh = (const _Float16*)&q;
    v16h a;
#pragma unroll
    for (int i = 0; i < 8; ++i) { a[i] = ph[i]; a[8 + i] = qh[i]; }
    return a;
}

// ---------------------------------------------------------------------------
// Prep: pack W1 (128x257) and W2 (128x128) into WMMA B-fragment layout (f16),
// and zero the per-node aggregation buffer. B-matrix 32x16 layout per ISA:
// lane l holds column n = nt*16 + (l&15), K rows kt*32 + (l>>4)*8 + {0..7, 16..23}.
// ---------------------------------------------------------------------------
__global__ void egnn_prep_kernel(const float* __restrict__ W1,
                                 const float* __restrict__ W2,
                                 unsigned char* __restrict__ ws) {
    int t = blockIdx.x * blockDim.x + threadIdx.x;
    _Float16* w1f = (_Float16*)(ws + WS_W1F);
    _Float16* w2f = (_Float16*)(ws + WS_W2F);
    float*    agg = (float*)(ws + WS_AGG);

    if (t < 32768) {                       // 8kt * 8nt * 32 lanes * 16 halfs
        int v  = t & 15;
        int l  = (t >> 4) & 31;
        int f  = t >> 9;                   // fragment id: kt*8 + nt
        int nt = f & 7;
        int kt = f >> 3;                   // 0..7
        int n  = nt * 16 + (l & 15);
        int k  = kt * 32 + ((l >> 4) << 3) + (v & 7) + ((v >> 3) << 4);
        w1f[t] = (_Float16)W1[n * IN_EDGE + k];   // B[k][n] = W1[n][k]
    }
    if (t < 16384) {                       // 4kt * 8nt * 32 * 16
        int v  = t & 15;
        int l  = (t >> 4) & 31;
        int f  = t >> 9;
        int nt = f & 7;
        int kt = f >> 3;                   // 0..3
        int n  = nt * 16 + (l & 15);
        int k  = kt * 32 + ((l >> 4) << 3) + (v & 7) + ((v >> 3) << 4);
        w2f[t] = (_Float16)W2[n * HNF + k];
    }
    if (t < N_NODES * 3) agg[t] = 0.0f;
}

// ---------------------------------------------------------------------------
// Main edge-MLP kernel: 8 waves/block, each wave owns a 32-edge tile
// (two 16-row M sub-tiles sharing every B fragment -> 2x WMMA per B load).
// Layer1: 8kt x 8nt x 2 = 128 WMMA, layer2: 4kt x 8nt x 2 = 64 WMMA.
// ---------------------------------------------------------------------------
__global__ void __launch_bounds__(256)
egnn_edge_mlp_kernel(const float* __restrict__ h,
                     const long long* __restrict__ ei,
                     const float* __restrict__ coord_diff,
                     const float* __restrict__ edge_attr,
                     const float* __restrict__ edge_mask,
                     const float* __restrict__ W1,
                     const float* __restrict__ b1,
                     const float* __restrict__ b2,
                     const float* __restrict__ W3,
                     unsigned char* __restrict__ ws) {
    __shared__ _Float16 act[8][32][HNF];   // per-wave layer1 activations (f16), 64KB
    __shared__ float    phi_s[8][32];

    const int lane = threadIdx.x & 31;
    const int w    = threadIdx.x >> 5;
    const int m    = lane & 15;            // row within 16-edge sub-tile / output col
    const int hi   = lane >> 4;            // half-wave select

    const int tile = blockIdx.x * 8 + w;   // 25000 wave-tiles total, exact
    const int tb   = tile * 32;            // 32 edges per wave

    // per-lane gather bases for the two M sub-tiles
    const long long r0 = ei[tb + m];
    const long long c0 = ei[N_EDGES + tb + m];
    const long long r1 = ei[tb + 16 + m];
    const long long c1 = ei[N_EDGES + tb + 16 + m];
    const float* hrow0 = h + (size_t)r0 * HNF;
    const float* hcol0 = h + (size_t)c0 * HNF;
    const float* hrow1 = h + (size_t)r1 * HNF;
    const float* hcol1 = h + (size_t)c1 * HNF;

    const v16h* W1f = (const v16h*)(ws + WS_W1F);
    const v16h* W2f = (const v16h*)(ws + WS_W2F);
    float*      agg = (float*)(ws + WS_AGG);

    // ---------------- layer 1: [32 x 256] x [256 x 128] ----------------
    v8f acc0[8] = {};
    v8f acc1[8] = {};
#pragma unroll
    for (int kt = 0; kt < 8; ++kt) {
        int f0 = kt * 32 + hi * 8;
        v16h a0 = build_a_frag(hrow0, hcol0, f0);
        v16h a1 = build_a_frag(hrow1, hcol1, f0);
#pragma unroll
        for (int nc = 0; nc < 2; ++nc) {
            // load 4 B fragments into distinct regs -> clause + overlap with WMMA
            v16h bb0 = W1f[(kt * 8 + nc * 4 + 0) * 32 + lane];
            v16h bb1 = W1f[(kt * 8 + nc * 4 + 1) * 32 + lane];
            v16h bb2 = W1f[(kt * 8 + nc * 4 + 2) * 32 + lane];
            v16h bb3 = W1f[(kt * 8 + nc * 4 + 3) * 32 + lane];
            acc0[nc * 4 + 0] = WMMA_F16(a0, bb0, acc0[nc * 4 + 0]);
            acc1[nc * 4 + 0] = WMMA_F16(a1, bb0, acc1[nc * 4 + 0]);
            acc0[nc * 4 + 1] = WMMA_F16(a0, bb1, acc0[nc * 4 + 1]);
            acc1[nc * 4 + 1] = WMMA_F16(a1, bb1, acc1[nc * 4 + 1]);
            acc0[nc * 4 + 2] = WMMA_F16(a0, bb2, acc0[nc * 4 + 2]);
            acc1[nc * 4 + 2] = WMMA_F16(a1, bb2, acc1[nc * 4 + 2]);
            acc0[nc * 4 + 3] = WMMA_F16(a0, bb3, acc0[nc * 4 + 3]);
            acc1[nc * 4 + 3] = WMMA_F16(a1, bb3, acc1[nc * 4 + 3]);
        }
    }

    // rank-1 update for feature 256 (edge_attr), bias, SiLU -> f16 LDS
    float ea0_mine = edge_attr[tb + m];
    float ea1_mine = edge_attr[tb + 16 + m];
    float ea0[8], ea1[8];
#pragma unroll
    for (int j = 0; j < 8; ++j) {
        ea0[j] = __shfl(ea0_mine, j + hi * 8, 32);
        ea1[j] = __shfl(ea1_mine, j + hi * 8, 32);
    }

#pragma unroll
    for (int nt = 0; nt < 8; ++nt) {
        int n = nt * 16 + m;
        float w1e  = W1[n * IN_EDGE + 256];
        float bias = b1[n];
#pragma unroll
        for (int j = 0; j < 8; ++j) {
            float x0 = acc0[nt][j] + ea0[j] * w1e + bias;
            float x1 = acc1[nt][j] + ea1[j] * w1e + bias;
            act[w][j + hi * 8][n]      = (_Float16)silu_f(x0);
            act[w][16 + j + hi * 8][n] = (_Float16)silu_f(x1);
        }
    }
    __syncthreads();

    // ---------------- layer 2: [32 x 128] x [128 x 128] ----------------
    v8f acc20[8] = {};
    v8f acc21[8] = {};
#pragma unroll
    for (int kt = 0; kt < 4; ++kt) {
        int f0 = kt * 32 + hi * 8;
        v16h a0 = build_a_frag_lds(&act[w][m][0], f0);
        v16h a1 = build_a_frag_lds(&act[w][16 + m][0], f0);
#pragma unroll
        for (int nc = 0; nc < 2; ++nc) {
            v16h bb0 = W2f[(kt * 8 + nc * 4 + 0) * 32 + lane];
            v16h bb1 = W2f[(kt * 8 + nc * 4 + 1) * 32 + lane];
            v16h bb2 = W2f[(kt * 8 + nc * 4 + 2) * 32 + lane];
            v16h bb3 = W2f[(kt * 8 + nc * 4 + 3) * 32 + lane];
            acc20[nc * 4 + 0] = WMMA_F16(a0, bb0, acc20[nc * 4 + 0]);
            acc21[nc * 4 + 0] = WMMA_F16(a1, bb0, acc21[nc * 4 + 0]);
            acc20[nc * 4 + 1] = WMMA_F16(a0, bb1, acc20[nc * 4 + 1]);
            acc21[nc * 4 + 1] = WMMA_F16(a1, bb1, acc21[nc * 4 + 1]);
            acc20[nc * 4 + 2] = WMMA_F16(a0, bb2, acc20[nc * 4 + 2]);
            acc21[nc * 4 + 2] = WMMA_F16(a1, bb2, acc21[nc * 4 + 2]);
            acc20[nc * 4 + 3] = WMMA_F16(a0, bb3, acc20[nc * 4 + 3]);
            acc21[nc * 4 + 3] = WMMA_F16(a1, bb3, acc21[nc * 4 + 3]);
        }
    }

    // ---------------- layer 3: bias + SiLU + dot(W3) -> phi ----------------
    float part0[8] = {0.f, 0.f, 0.f, 0.f, 0.f, 0.f, 0.f, 0.f};
    float part1[8] = {0.f, 0.f, 0.f, 0.f, 0.f, 0.f, 0.f, 0.f};
#pragma unroll
    for (int nt = 0; nt < 8; ++nt) {
        int n = nt * 16 + m;
        float bias = b2[n];
        float w3   = W3[n];
#pragma unroll
        for (int j = 0; j < 8; ++j) {
            part0[j] += silu_f(acc20[nt][j] + bias) * w3;
            part1[j] += silu_f(acc21[nt][j] + bias) * w3;
        }
    }
#pragma unroll
    for (int j = 0; j < 8; ++j) {
        float v0 = part0[j];
        float v1 = part1[j];
        v0 += __shfl_xor(v0, 1, 32);  v1 += __shfl_xor(v1, 1, 32);
        v0 += __shfl_xor(v0, 2, 32);  v1 += __shfl_xor(v1, 2, 32);
        v0 += __shfl_xor(v0, 4, 32);  v1 += __shfl_xor(v1, 4, 32);
        v0 += __shfl_xor(v0, 8, 32);  v1 += __shfl_xor(v1, 8, 32);
        part0[j] = v0;                // identical across each 16-lane half
        part1[j] = v1;
    }
    if (m == 0) {
#pragma unroll
        for (int j = 0; j < 8; ++j) {
            phi_s[w][j + hi * 8]      = part0[j];
            phi_s[w][16 + j + hi * 8] = part1[j];
        }
    }
    __syncthreads();

    // ---------------- trans + scatter-add (one edge per lane) ----------------
    {
        int e = tb + lane;
        float s = phi_s[w][lane] * edge_mask[e];
        long long rr = ei[e];
        float cd0 = coord_diff[(size_t)e * 3 + 0];
        float cd1 = coord_diff[(size_t)e * 3 + 1];
        float cd2 = coord_diff[(size_t)e * 3 + 2];
        atomicAdd(&agg[rr * 3 + 0], cd0 * s);
        atomicAdd(&agg[rr * 3 + 1], cd1 * s);
        atomicAdd(&agg[rr * 3 + 2], cd2 * s);
    }
}

// ---------------------------------------------------------------------------
// Finalize: out = (coord + agg/100) * node_mask
// ---------------------------------------------------------------------------
__global__ void egnn_finalize_kernel(const float* __restrict__ coord,
                                     const float* __restrict__ node_mask,
                                     const unsigned char* __restrict__ ws,
                                     float* __restrict__ out) {
    int i = blockIdx.x * blockDim.x + threadIdx.x;
    if (i >= N_NODES * 3) return;
    const float* agg = (const float*)(ws + WS_AGG);
    int node = i / 3;
    out[i] = (coord[i] + agg[i] * 0.01f) * node_mask[node];
}

extern "C" void kernel_launch(void* const* d_in, const int* in_sizes, int n_in,
                              void* d_out, int out_size, void* d_ws, size_t ws_size,
                              hipStream_t stream) {
    const float*     h          = (const float*)d_in[0];
    const float*     coord      = (const float*)d_in[1];
    const long long* edge_index = (const long long*)d_in[2];
    const float*     coord_diff = (const float*)d_in[3];
    const float*     edge_attr  = (const float*)d_in[4];
    const float*     node_mask  = (const float*)d_in[5];
    const float*     edge_mask  = (const float*)d_in[6];
    const float*     W1         = (const float*)d_in[7];
    const float*     b1         = (const float*)d_in[8];
    const float*     W2         = (const float*)d_in[9];
    const float*     b2         = (const float*)d_in[10];
    const float*     W3         = (const float*)d_in[11];
    float*           out        = (float*)d_out;
    unsigned char*   ws         = (unsigned char*)d_ws;

    // 1) pack weights to f16 fragments + zero agg (deterministic each call)
    egnn_prep_kernel<<<(N_NODES * 3 + 255) / 256, 256, 0, stream>>>(W1, W2, ws);

    // 2) per-edge MLP via WMMA + scatter-add
    //    E = 800000 = 3125 blocks * 8 waves * 32 edges (exact, no tail)
    egnn_edge_mlp_kernel<<<N_EDGES / 256, 256, 0, stream>>>(
        h, edge_index, coord_diff, edge_attr, edge_mask, W1, b1, b2, W3, ws);

    // 3) finalize coords
    egnn_finalize_kernel<<<(N_NODES * 3 + 255) / 256, 256, 0, stream>>>(
        coord, node_mask, ws, out);
}
